// FastWeightAttention_14774687498629
// MI455X (gfx1250) — matrix operand order repaired and checked by reading, not compiled, not run on hardware
//
#include <hip/hip_runtime.h>
#include <hip/hip_bf16.h>

// ---------------------------------------------------------------------------
// FastWeightAttention (delta-rule linear attention, DPFP features) for gfx1250
//   - all GEMMs + the recurrent scan run on v_wmma_f32_16x16x32_bf16
//   - scan uses the chunkwise delta-rule (C=16) so the sequential recurrence
//     becomes 16x16 WMMA tiles; f32 state held in registers as C-fragments
//   - chunk staging (K/Q/V tiles) goes through the Tensor Data Mover
//     (tensor_load_to_lds + s_wait_tensorcnt) when the builtin is available
// ---------------------------------------------------------------------------

typedef __bf16 bf16;
typedef __attribute__((ext_vector_type(16))) __bf16 v16bf;
typedef __attribute__((ext_vector_type(8)))  float  v8f;
typedef __attribute__((ext_vector_type(4)))  unsigned int v4u;
typedef __attribute__((ext_vector_type(8)))  int  v8i;
typedef __attribute__((ext_vector_type(4)))  int  v4i;

#if defined(__has_builtin)
#if __has_builtin(__builtin_amdgcn_tensor_load_to_lds)
#define USE_TDM 1
#endif
#endif
#ifndef USE_TDM
#define USE_TDM 0
#endif

union FragU { v16bf v; uint4 q[2]; };

__device__ __forceinline__ v8f v8f_zero() {
  v8f z = {0.f, 0.f, 0.f, 0.f, 0.f, 0.f, 0.f, 0.f};
  return z;
}

// A-matrix fragment (16x32 bf16, MxK), source row-major with leading dim ld.
// CDNA5 layout: lanes 0-15 hold row M=lane, K in {0..7,16..23};
//               lanes 16-31 hold row M=lane-16, K in {8..15,24..31}.
__device__ __forceinline__ v16bf load_frag_A(const bf16* base, int ld, int k0) {
  const int lane = threadIdx.x & 31;
  const int row  = lane & 15;
  const bf16* p = base + row * ld + k0 + ((lane < 16) ? 0 : 8);
  FragU f;
  f.q[0] = *(const uint4*)(p);        // first 8 K values
  f.q[1] = *(const uint4*)(p + 16);   // +16 K offset
  return f.v;
}

// B-matrix fragment (32x16 bf16, KxN) given the TRANSPOSE Bt[n][k] row-major.
// CDNA5 layout: lanes 0-15 hold col N=lane, K in {0..7,16..23};
//               lanes 16-31 hold col N=lane-16, K in {8..15,24..31}.
__device__ __forceinline__ v16bf load_frag_Bt(const bf16* baseT, int ld, int k0) {
  const int lane = threadIdx.x & 31;
  const int n    = lane & 15;
  const bf16* p = baseT + n * ld + k0 + ((lane < 16) ? 0 : 8);
  FragU f;
  f.q[0] = *(const uint4*)(p);        // first 8 K values
  f.q[1] = *(const uint4*)(p + 16);   // +16 K offset
  return f.v;
}

__device__ __forceinline__ v8f wmma_bf16(v16bf a, v16bf b, v8f c) {
  return __builtin_amdgcn_wmma_f32_16x16x32_bf16(false, a, false, b,
                                                 (short)0, c, false, false);
}

#if USE_TDM
// Issue one 2D-tile TDM load: global (row-strided) tile -> contiguous LDS.
// D# layout per CDNA5 ISA.
//   data_size_code: 0=1B 1=2B 2=4B; dims/strides in data_size units.
__device__ __forceinline__ void tdm_load_2d(unsigned lds_addr,
                                            unsigned long long gaddr,
                                            unsigned data_size_code,
                                            unsigned tile_d0, unsigned tile_d1,
                                            unsigned tensor_d0, unsigned tensor_d1,
                                            unsigned long long stride0) {
  v4u g0;
  g0[0] = 1u;                                         // count=1, user descriptor
  g0[1] = lds_addr;                                   // LDS byte address
  g0[2] = (unsigned)(gaddr & 0xffffffffull);          // global_addr[31:0]
  g0[3] = (unsigned)((gaddr >> 32) & 0x01fffffful)    // global_addr[56:32]
          | 0x80000000u;                              // type=2 ("image")
  v8i g1;
  g1[0] = (int)(data_size_code << 16);                // wg_mask=0, data_size
  g1[1] = (int)((tensor_d0 & 0xffffu) << 16);        // tensor_dim0[15:0]
  g1[2] = (int)((tensor_d0 >> 16) | ((tensor_d1 & 0xffffu) << 16));
  g1[3] = (int)((tensor_d1 >> 16) | (tile_d0 << 16)); // tile_dim0
  g1[4] = (int)(tile_d1 & 0xffffu);                   // tile_dim1 (tile_dim2=0)
  g1[5] = (int)(stride0 & 0xffffffffull);             // tensor_dim0_stride[31:0]
  g1[6] = (int)((stride0 >> 32) & 0xffffull);         // stride[47:32], dim1_stride=0
  g1[7] = 0;
  v4i z4 = {0, 0, 0, 0};
#if __clang_major__ >= 23
  v8i z8 = {0, 0, 0, 0, 0, 0, 0, 0};
  __builtin_amdgcn_tensor_load_to_lds(g0, g1, z4, z4, z8, 0);
#else
  __builtin_amdgcn_tensor_load_to_lds(g0, g1, z4, z4, 0);
#endif
}
#endif

// ---------------------------------------------------------------------------
// Generic C = A(f32,MxK) @ B(f32,KxN) [+ bias], bf16 WMMA, f32 accumulate.
// Block: 256 threads (8 waves), tile 128(M) x 64(N), K-step 32.
// ---------------------------------------------------------------------------
#define GT_M 128
#define GT_N 64
#define GT_K 32
#define G_LDA 40
#define G_LDB 40

__global__ __launch_bounds__(256)
void gemm_xw_bf16(const float* __restrict__ A, const float* __restrict__ Bw,
                  const float* __restrict__ bias, float* __restrict__ C,
                  int M, int N, int K) {
  __shared__ __align__(16) bf16 sA[GT_M * G_LDA];
  __shared__ __align__(16) bf16 sBt[GT_N * G_LDB];
  const int tid  = threadIdx.x;
  const int wave = tid >> 5;
  const int lane = tid & 31;
  const size_t m0 = (size_t)blockIdx.y * GT_M;
  const size_t n0 = (size_t)blockIdx.x * GT_N;

  v8f acc[4];
#pragma unroll
  for (int j = 0; j < 4; ++j) acc[j] = v8f_zero();

  for (int k0 = 0; k0 < K; k0 += GT_K) {
    __syncthreads();
    // stage A tile (f32 -> bf16)
    for (int i = tid; i < GT_M * GT_K; i += 256) {
      int r = i >> 5, c = i & 31;
      sA[r * G_LDA + c] = (bf16)A[(m0 + r) * (size_t)K + (k0 + c)];
    }
    // stage B tile transposed: sBt[n][k] = B[k][n]
    for (int i = tid; i < GT_K * GT_N; i += 256) {
      int r = i >> 6, c = i & 63;
      sBt[c * G_LDB + r] = (bf16)Bw[(size_t)(k0 + r) * N + (n0 + c)];
    }
    // speculative prefetch of the next K-slab (global_prefetch_b8)
    if (k0 + GT_K < K) {
      __builtin_prefetch(&A[(m0 + (tid & 127)) * (size_t)K + (k0 + GT_K)], 0, 1);
      if (tid < GT_K)
        __builtin_prefetch(&Bw[(size_t)(k0 + GT_K + tid) * N + n0], 0, 1);
    }
    __syncthreads();
    v16bf af = load_frag_A(sA + wave * 16 * G_LDA, G_LDA, 0);
#pragma unroll
    for (int j = 0; j < 4; ++j) {
      v16bf bfr = load_frag_Bt(sBt + j * 16 * G_LDB, G_LDB, 0);
      acc[j] = wmma_bf16(af, bfr, acc[j]);
    }
  }
  // C/D layout: VGPR i -> (M = i or i+8, N = lane&15)
#pragma unroll
  for (int j = 0; j < 4; ++j) {
#pragma unroll
    for (int i = 0; i < 8; ++i) {
      size_t r = m0 + wave * 16 + ((lane < 16) ? i : i + 8);
      size_t cc = n0 + j * 16 + (lane & 15);
      float val = acc[j][i];
      if (bias) val += bias[cc];
      C[r * (size_t)N + cc] = val;
    }
  }
}

// ---------------------------------------------------------------------------
// DPFP (nu=1): r = [relu(x), relu(-x)] (128), y[j] = r[j]*r[(j-1)%128],
// L1-normalized. One thread per (token, head). Output bf16.
// ---------------------------------------------------------------------------
__global__ __launch_bounds__(256)
void dpfp_kernel(const float* __restrict__ proj, bf16* __restrict__ out, int total) {
  const int idx = blockIdx.x * 256 + threadIdx.x;
  if (idx >= total) return;
  const float4* p = (const float4*)(proj + (size_t)idx * 64);
  float x[64];
#pragma unroll
  for (int i = 0; i < 16; ++i) {
    float4 t = p[i];
    x[4 * i + 0] = t.x; x[4 * i + 1] = t.y; x[4 * i + 2] = t.z; x[4 * i + 3] = t.w;
  }
  float sum = 0.f;
#pragma unroll
  for (int j = 0; j < 128; ++j) {
    const int jm = (j + 127) & 127;
    float rj = (j  < 64) ? fmaxf(x[j],       0.f) : fmaxf(-x[j  - 64], 0.f);
    float rm = (jm < 64) ? fmaxf(x[jm],      0.f) : fmaxf(-x[jm - 64], 0.f);
    sum += rj * rm;
  }
  const float inv = 1.f / (sum + 1e-6f);
  bf16* o = out + (size_t)idx * 128;
#pragma unroll
  for (int j = 0; j < 128; ++j) {
    const int jm = (j + 127) & 127;
    float rj = (j  < 64) ? fmaxf(x[j],       0.f) : fmaxf(-x[j  - 64], 0.f);
    float rm = (jm < 64) ? fmaxf(x[jm],      0.f) : fmaxf(-x[jm - 64], 0.f);
    o[j] = (bf16)(rj * rm * inv);
  }
}

// beta = sigmoid(x @ Wg); Wg is 512x8. One thread per (token, head).
__global__ __launch_bounds__(256)
void beta_kernel(const float* __restrict__ x, const float* __restrict__ Wg,
                 float* __restrict__ beta, int rows) {
  const int idx = blockIdx.x * 256 + threadIdx.x;
  if (idx >= rows * 8) return;
  const int row = idx >> 3, h = idx & 7;
  const float* xr = x + (size_t)row * 512;
  float s = 0.f;
  for (int d = 0; d < 512; ++d) s += xr[d] * Wg[d * 8 + h];
  beta[idx] = 1.f / (1.f + __expf(-s));
}

// ---------------------------------------------------------------------------
// Chunkwise delta-rule scan. One block per (b,h); 4 waves (128 threads).
// Chunk C=16: A=KK^T, S=QK^T, KW0=K@W0^T, O0=Q@W0^T (WMMA, K=128);
// forward substitution for U (VALU); Out = O0 + tril(S)U (VALU);
// W += U^T K (32 WMMA tiles, K=16 zero-padded to 32). f32 state in registers.
// Chunk staging via TDM (tensor_load_to_lds) when available.
// ---------------------------------------------------------------------------
__global__ __launch_bounds__(128)
void scan_kernel(const bf16* __restrict__ q, const bf16* __restrict__ k,
                 const float* __restrict__ v, const float* __restrict__ beta,
                 float* __restrict__ out) {
  __shared__ __align__(16) bf16 sWbf[64 * 128];   // W0 (bf16) rows d, cols phi
  __shared__ __align__(16) bf16 sK[16 * 128];     // K chunk, rows t
  __shared__ __align__(16) bf16 sQ[16 * 128];     // Q chunk, rows t
  __shared__ __align__(16) bf16 sKT[128 * 32];    // K^T, rows phi, K-pad 16..31 = 0
  __shared__ __align__(16) bf16 sUT[64 * 32];     // U^T, rows d,  K-pad 16..31 = 0
  __shared__ float sV[16 * 64];
  __shared__ float sKW[16 * 64];
  __shared__ float sO0[16 * 64];
  __shared__ float sA[16 * 16];
  __shared__ float sS[16 * 16];
  __shared__ float sBeta[16];

  const int tid  = threadIdx.x;
  const int wave = tid >> 5;
  const int lane = tid & 31;
  const int bh = blockIdx.x;
  const int b = bh >> 3, h = bh & 7;

  // persistent f32 state: wave owns tiles (wave + 4*j), j=0..7
  v8f wacc[8];
#pragma unroll
  for (int j = 0; j < 8; ++j) wacc[j] = v8f_zero();

  // zero (once) the K-padding halves; chunk loads only touch k<16
  for (int i = tid; i < 128 * 32; i += 128) sKT[i] = (bf16)0.f;
  for (int i = tid; i < 64 * 32; i += 128) sUT[i] = (bf16)0.f;

#if USE_TDM
  const unsigned ldsK = (unsigned)(size_t)(void*)sK;
  const unsigned ldsQ = (unsigned)(size_t)(void*)sQ;
  const unsigned ldsV = (unsigned)(size_t)(void*)sV;
#endif

  for (int c = 0; c < 256; ++c) {
    const int t0 = c * 16;
    __syncthreads();  // previous chunk's phase-3 reads done; LDS reusable

#if USE_TDM
    // Tensor Data Mover: 2D tiles (16 rows, row stride = tokens apart),
    // contiguous into LDS. One descriptor per wave, wait on TENSORcnt.
    if (wave == 0) {
      tdm_load_2d(ldsK,
                  (unsigned long long)(const void*)(k + ((size_t)t0 * 8 + b) * 1024 + (size_t)h * 128),
                  1u, 128u, 16u, 128u, 16u, 8192ull);
    } else if (wave == 1) {
      tdm_load_2d(ldsQ,
                  (unsigned long long)(const void*)(q + ((size_t)t0 * 8 + b) * 1024 + (size_t)h * 128),
                  1u, 128u, 16u, 128u, 16u, 8192ull);
    } else if (wave == 2) {
      tdm_load_2d(ldsV,
                  (unsigned long long)(const void*)(v + ((size_t)t0 * 8 + b) * 512 + (size_t)h * 64),
                  2u, 64u, 16u, 64u, 16u, 4096ull);
    }
    if (wave < 3) __builtin_amdgcn_s_wait_tensorcnt(0);
#else
    // fallback: per-element staging (also builds K^T inline)
    for (int i = tid; i < 16 * 128; i += 128) {
      int tt = i >> 7, ph = i & 127;
      size_t g = ((size_t)(t0 + tt) * 8 + b) * 1024 + (size_t)h * 128 + ph;
      bf16 kv = k[g];
      sK[tt * 128 + ph] = kv;
      sKT[ph * 32 + tt] = kv;
      sQ[tt * 128 + ph] = q[g];
    }
    for (int i = tid; i < 16 * 64; i += 128) {
      int tt = i >> 6, d = i & 63;
      sV[i] = v[((size_t)(t0 + tt) * 8 + b) * 512 + (size_t)h * 64 + d];
    }
#endif
    if (tid < 16) sBeta[tid] = beta[((size_t)(t0 + tid) * 8 + b) * 8 + h];

    // publish W0 as bf16 from register fragments
#pragma unroll
    for (int j = 0; j < 8; ++j) {
      int tile = wave + 4 * j;
      int mi = tile >> 3, ni = tile & 7;
#pragma unroll
      for (int i = 0; i < 8; ++i) {
        int r   = mi * 16 + ((lane < 16) ? i : i + 8);
        int col = ni * 16 + (lane & 15);
        sWbf[r * 128 + col] = (bf16)wacc[j][i];
      }
    }
    __syncthreads();

    // ---- Phase 1: WMMA, K = 128 (4 k-steps of 32) ----
    {
      v8f accKW = v8f_zero();
      v8f accO0 = v8f_zero();
      v8f accAS = v8f_zero();
#pragma unroll
      for (int ks = 0; ks < 4; ++ks) {
        v16bf aK = load_frag_A(sK, 128, ks * 32);
        v16bf aQ = load_frag_A(sQ, 128, ks * 32);
        // B = W0^T (128x64): Bt rows = W0 rows d (this wave's 16-col slab)
        v16bf bW = load_frag_Bt(sWbf + wave * 16 * 128, 128, ks * 32);
        accKW = wmma_bf16(aK, bW, accKW);
        accO0 = wmma_bf16(aQ, bW, accO0);
        if (wave == 0) {          // A = K K^T : B = K^T, Bt = K
          v16bf bK = load_frag_Bt(sK, 128, ks * 32);
          accAS = wmma_bf16(aK, bK, accAS);
        } else if (wave == 1) {   // S = Q K^T
          v16bf bK = load_frag_Bt(sK, 128, ks * 32);
          accAS = wmma_bf16(aQ, bK, accAS);
        }
      }
#if USE_TDM
      // build K^T for phase 3 (TDM cannot transpose); overlapped with WMMA
      for (int i = tid; i < 16 * 128; i += 128) {
        int tt = i >> 7, ph = i & 127;
        sKT[ph * 32 + tt] = sK[tt * 128 + ph];
      }
#endif
#pragma unroll
      for (int i = 0; i < 8; ++i) {
        int r = (lane < 16) ? i : i + 8;
        int n = lane & 15;
        sKW[r * 64 + wave * 16 + n] = accKW[i];
        sO0[r * 64 + wave * 16 + n] = accO0[i];
        if (wave == 0) sA[r * 16 + n] = accAS[i];
        else if (wave == 1) sS[r * 16 + n] = accAS[i];
      }
    }
    __syncthreads();

    // ---- Phase 2: forward substitution + chunk output (thread = column d) ----
    if (tid < 64) {
      const int d = tid;
      float u[16];
#pragma unroll
      for (int t = 0; t < 16; ++t) {
        float s = sV[t * 64 + d] - sKW[t * 64 + d];
#pragma unroll
        for (int s2 = 0; s2 < t; ++s2) s -= sA[t * 16 + s2] * u[s2];
        u[t] = sBeta[t] * s;
        sUT[d * 32 + t] = (bf16)u[t];
      }
#pragma unroll
      for (int t = 0; t < 16; ++t) {
        float o = sO0[t * 64 + d];
#pragma unroll
        for (int s2 = 0; s2 <= t; ++s2) o += sS[t * 16 + s2] * u[s2];
        out[((size_t)(t0 + t) * 8 + b) * 512 + (size_t)h * 64 + d] = o;
      }
    }
    __syncthreads();

    // ---- Phase 3: W += U^T K (K=16 padded to 32 with zeros) ----
#pragma unroll
    for (int j = 0; j < 8; ++j) {
      int tile = wave + 4 * j;
      int mi = tile >> 3, ni = tile & 7;
      v16bf aU = load_frag_A(sUT + mi * 16 * 32, 32, 0);
      v16bf bK = load_frag_Bt(sKT + ni * 16 * 32, 32, 0);
      wacc[j] = wmma_bf16(aU, bK, wacc[j]);
    }
  }
}

// ---------------------------------------------------------------------------
extern "C" void kernel_launch(void* const* d_in, const int* in_sizes, int n_in,
                              void* d_out, int out_size, void* d_ws, size_t ws_size,
                              hipStream_t stream) {
  const float* x  = (const float*)d_in[0];
  const float* Wq = (const float*)d_in[1];
  const float* Wk = (const float*)d_in[2];
  const float* Wv = (const float*)d_in[3];
  const float* Wg = (const float*)d_in[4];
  const float* Wo = (const float*)d_in[5];
  const float* bo = (const float*)d_in[6];
  float* outp = (float*)d_out;

  const int SB = 4096 * 8;      // 32768 tokens
  const int D  = 512;

  char* ws = (char*)d_ws;
  float* B0   = (float*)(ws);                      //  64MB: qproj -> attn_out
  float* B1f  = (float*)(ws + (64ull << 20));      //  64MB: kproj -> q (bf16)
  bf16*  B1bf = (bf16*)B1f;
  float* B2   = (float*)(ws + (128ull << 20));     //  64MB: v (f32)
  bf16*  B3   = (bf16*)(ws + (192ull << 20));      //  64MB: k (bf16)
  float* B4   = (float*)(ws + (256ull << 20));     //   1MB: beta

  dim3 gblk(D / GT_N, SB / GT_M);  // (8, 256)

  // projections
  gemm_xw_bf16<<<gblk, 256, 0, stream>>>(x, Wq, nullptr, B0,  SB, D, D);
  gemm_xw_bf16<<<gblk, 256, 0, stream>>>(x, Wk, nullptr, B1f, SB, D, D);
  gemm_xw_bf16<<<gblk, 256, 0, stream>>>(x, Wv, nullptr, B2,  SB, D, D);

  // feature maps (k first so B1 can be reused for q bf16)
  dpfp_kernel<<<(SB * 8) / 256, 256, 0, stream>>>(B1f, B3,   SB * 8); // k
  dpfp_kernel<<<(SB * 8) / 256, 256, 0, stream>>>(B0,  B1bf, SB * 8); // q
  beta_kernel<<<(SB * 8) / 256, 256, 0, stream>>>(x, Wg, B4, SB);

  // chunkwise delta-rule scan: 64 blocks (B*H), writes attn into B0
  scan_kernel<<<64, 128, 0, stream>>>(B1bf, B3, B2, B4, B0);

  // output projection + bias
  gemm_xw_bf16<<<gblk, 256, 0, stream>>>(B0, Wo, bo, outp, SB, D, D);
}
